// PolarCode_40604620816836
// MI455X (gfx1250) — 8-bit matrix operand order repaired and checked by reading, compiled, not run on hardware
//
#include <hip/hip_runtime.h>
#include <hip/hip_bf16.h>


#define BATCH 16384
#define KDIM  512
#define NDIM  1024
#define KBYTES 512
#define KWORDS 128
#define M_TILE 64
#define NSPLIT 4
#define CHUNKS_PER_WG (NDIM / 16 / NSPLIT)

typedef int v8i __attribute__((ext_vector_type(8)));

__device__ __forceinline__ unsigned int bit01(float f) {
    return (__float_as_uint(f) >> 23) & 1u;
}

__global__ __launch_bounds__(256) void pack_matrix_kernel(
    const float* __restrict__ m, unsigned int* __restrict__ Bp) {
    const int tid = blockIdx.x * 256 + threadIdx.x;
    const int n  = tid & (NDIM - 1);
    const int kw = tid >> 10;
    unsigned int p = 0;
    #pragma unroll
    for (int j = 0; j < 4; ++j)
        p |= bit01(m[(4 * kw + j) * NDIM + n]) << (8 * j);
    Bp[n * KWORDS + kw] = p;
}

__global__ __launch_bounds__(128) void gf2_wmma_gemm_kernel(
    const float* __restrict__ x, const unsigned int* __restrict__ Bp,
    int* __restrict__ out) {
    __shared__ unsigned int Als[M_TILE * KWORDS];
    __shared__ unsigned int Bls[16 * KWORDS];

    const int t    = threadIdx.x;
    const int lane = t & 31;
    const int wave = t >> 5;
    const int r0   = blockIdx.x * M_TILE;

    #pragma unroll 4
    for (int i = 0; i < M_TILE; ++i) {
        const float4 f = *(const float4*)(x + (size_t)(r0 + i) * KDIM + t * 4);
        unsigned int p =  bit01(f.x)
                       | (bit01(f.y) << 8)
                       | (bit01(f.z) << 16)
                       | (bit01(f.w) << 24);
        Als[i * KWORDS + t] = p;
    }

    const int  nloc = lane & 15;
    const int  rowA = (wave << 4) + nloc;
    const int  hA   = (lane < 16) ? 0 : 16;
    const int  hB   = (lane < 16) ? 0 : 16;
    const unsigned char* Ab = (const unsigned char*)Als;
    const unsigned char* Bb = (const unsigned char*)Bls;

    const int c0 = blockIdx.y * CHUNKS_PER_WG;
    for (int c = c0; c < c0 + CHUNKS_PER_WG; ++c) {
        __syncthreads();
        #pragma unroll
        for (int i = 0; i < 16; ++i)
            Bls[i * KWORDS + t] = Bp[c * (16 * KWORDS) + i * KWORDS + t];
        __syncthreads();

        v8i acc = {0, 0, 0, 0, 0, 0, 0, 0};
        #pragma unroll
        for (int kc = 0; kc < 8; ++kc) {
            const int abase = rowA * KBYTES + kc * 64 + hA;
            uint2 p0 = *(const uint2*)(Ab + abase);
            uint2 p1 = *(const uint2*)(Ab + abase + 8);
            uint2 p2 = *(const uint2*)(Ab + abase + 32);
            uint2 p3 = *(const uint2*)(Ab + abase + 40);
            v8i a = { (int)p0.x, (int)p0.y, (int)p1.x, (int)p1.y,
                      (int)p2.x, (int)p2.y, (int)p3.x, (int)p3.y };
            const int bbase = nloc * KBYTES + kc * 64 + hB;
            uint4 q0 = *(const uint4*)(Bb + bbase);
            uint4 q1 = *(const uint4*)(Bb + bbase + 32);
            v8i b = { (int)q0.x, (int)q0.y, (int)q0.z, (int)q0.w,
                      (int)q1.x, (int)q1.y, (int)q1.z, (int)q1.w };
            acc = __builtin_amdgcn_wmma_i32_16x16x64_iu8(false, a, false, b,
                                                         acc, false, false);
        }

        const int colg  = c * 16 + nloc;
        const int rbase = r0 + (wave << 4) + ((lane < 16) ? 0 : 8);
        #pragma unroll
        for (int i = 0; i < 8; ++i)
            out[(size_t)(rbase + i) * NDIM + colg] = acc[i] & 1;
    }
}

extern "C" void kernel_launch(void* const* d_in, const int* in_sizes, int n_in,
                              void* d_out, int out_size, void* d_ws, size_t ws_size,
                              hipStream_t stream) {
    const float* x = (const float*)d_in[0];
    const float* g = (const float*)d_in[1];
    int* out = (int*)d_out;
    unsigned int* Bp = (unsigned int*)d_ws;

    pack_matrix_kernel<<<512, 256, 0, stream>>>(g, Bp);
    dim3 grid(BATCH / M_TILE, NSPLIT);
    gf2_wmma_gemm_kernel<<<grid, 128, 0, stream>>>(x, Bp, out);
}
